// IConv_16844861735554
// MI455X (gfx1250) — matrix operand order repaired and checked by reading, not compiled, not run on hardware
//
#include <hip/hip_runtime.h>

// ---------------------------------------------------------------------------
// Fused IConv for MI455X (gfx1250, wave32, WMMA f32_16x16x32_f16)
//
//   stage A: kern2[r=o*9+k][d] = relu( sum_l w[o,k,l] * mhc[n,l,d] )   (LDS)
//   stage B: out[n,o,t]        = sum_{k,d} pep[n,t+k,d] * kern2[o*9+k][d]
//
// One workgroup = 8 wave32 = one (n, 32-wide o-block). Intermediate kernel
// tensor lives only in LDS (saves 2.4 GB of HBM round-trip vs naive).
// Stage B is split-K across all 8 waves (18 WMMA each) + LDS pair-reduce,
// balancing the WMMA critical path at 36+18 per wave.
// ---------------------------------------------------------------------------

typedef __attribute__((ext_vector_type(16))) _Float16 v16h;
typedef __attribute__((ext_vector_type(8)))  _Float16 v8h;
typedef __attribute__((ext_vector_type(4)))  _Float16 v4h;
typedef __attribute__((ext_vector_type(8)))  float    v8f;

#define WMMA16(A, B, C) \
  __builtin_amdgcn_wmma_f32_16x16x32_f16(false, (A), false, (B), (short)0, (C), false, false)

// Problem constants
#define CN    512
#define CL    40
#define CD    128
#define CO    256
#define CK    9
#define CMHC  34
#define CT    32          // (40 - 9)/1 + 1
#define OB    32          // o per block
#define RB    (OB * CK)   // 288 rows of the per-block kernel tensor
#define LP    64          // MHC padded to multiple of 32

// LDS layout (halfs)
#define SM_W_OFF   0                        // smW  [RB][LP]      288*64 = 18432
#define SM_BT_OFF  (SM_W_OFF + RB * LP)     // smBt [CD][LP]      128*64 =  8192
#define SM_P_OFF   (SM_BT_OFF + CD * LP)    // smP  [CL][CD]      40*128 =  5120
#define SM_K_OFF   (SM_P_OFF + CL * CD)     // smK  [RB][CD]     288*128 = 36864
#define SM_TOTAL   (SM_K_OFF + RB * CD)     // 68608 halfs = 137216 bytes
// Stage-B partial-accumulator scratch: 8 waves * 32 lanes * 8 f32 = 8 KB.
// Reuses the smW region (dead after the stage-A -> stage-B barrier).

// A-matrix fragment (16x32 f16):
//   lane m = lane&15 is the row; lane group (lane>>4) selects K 0..15 / 16..31;
//   elements 0..15 -> K = kb+0..15 (matches frag_b).
// p points at &mat[row][kcol0 + (lane>>4)*16]; rows contiguous in LDS.
__device__ __forceinline__ v16h frag_a(const _Float16* p) {
  v8h lo = *(const v8h*)(p);
  v8h hi = *(const v8h*)(p + 8);
  v16h r;
#pragma unroll
  for (int i = 0; i < 8; ++i) { r[i] = lo[i]; r[i + 8] = hi[i]; }
  return r;
}

// B-matrix fragment (32x16 f16): lane n = lane&15 is the column; lane group
// selects K 0..15 / 16..31; 16 contiguous halfs per lane.
__device__ __forceinline__ v16h frag_b(const _Float16* p) {
  return *(const v16h*)(p);
}

__global__ __launch_bounds__(256)
void iconv_fused_kernel(const float* __restrict__ pep,   // [N][L][D]
                        const float* __restrict__ mhc,   // [N][MHC][D]
                        const float* __restrict__ wgt,   // [O][K][MHC]
                        const float* __restrict__ bias,  // [O]
                        float* __restrict__ out)         // [N][O][T]
{
  extern __shared__ __attribute__((aligned(32))) _Float16 smem[];
  _Float16* smW  = smem + SM_W_OFF;   // A source, stage A   [RB][LP]
  _Float16* smBt = smem + SM_BT_OFF;  // B source, stage A   [CD][LP] (mhc^T)
  _Float16* smP  = smem + SM_P_OFF;   // B source, stage B   [CL][CD]
  _Float16* smK  = smem + SM_K_OFF;   // C of stage A / A of stage B [RB][CD]
  float*    scr  = (float*)smem;      // stage-B partials (aliases smW region)

  const int n     = blockIdx.x >> 3;      // sample
  const int ob    = blockIdx.x & 7;       // o-block
  const int obase = ob * OB;
  const int tid   = threadIdx.x;
  const int lane  = tid & 31;
  const int wave  = tid >> 5;
  const int lm    = lane & 15;            // row/col within tile
  const int lg    = lane >> 4;            // lane group (0/1)

  // ---------------- Phase 0: stage + convert fp32 -> f16 into LDS ----------
  // weight rows (o*K+k) are contiguous for this o-block: 288 rows x 34 cols.
  {
    const float* gW = wgt + (size_t)(obase * CK) * CMHC;
#pragma unroll 4
    for (int i = tid; i < RB * LP; i += 256) {
      const int r = i >> 6, c = i & 63;
      smW[i] = (c < CMHC) ? (_Float16)gW[r * CMHC + c] : (_Float16)0.0f;
    }
    // mhc -> transposed LDS tile smBt[d][l] (coalesced global read)
    const float* gM = mhc + (size_t)n * CMHC * CD;
#pragma unroll 4
    for (int i = tid; i < CMHC * CD; i += 256) {
      const int l = i >> 7, d = i & 127;
      smBt[d * LP + l] = (_Float16)gM[i];
    }
#pragma unroll 4
    for (int i = tid; i < (LP - CMHC) * CD; i += 256) {   // zero l = 34..63
      const int l = CMHC + (i >> 7), d = i & 127;
      smBt[d * LP + l] = (_Float16)0.0f;
    }
    // peptide: float4 loads -> packed 4 x f16 ds_store_b64
    const float4* gP4 = (const float4*)(pep + (size_t)n * CL * CD);
#pragma unroll 2
    for (int i = tid; i < (CL * CD) / 4; i += 256) {
      const float4 v = gP4[i];
      v4h h; h[0] = (_Float16)v.x; h[1] = (_Float16)v.y;
             h[2] = (_Float16)v.z; h[3] = (_Float16)v.w;
      *(v4h*)&smP[i * 4] = h;
    }
  }
  __syncthreads();

  // ---------------- Stage A: smK = relu( smW(288x34) @ mhc(34x128) ) -------
  // 18 M-tiles x 8 N-tiles; each wave owns one N-tile (d-column strip) so its
  // two B fragments (l = 0..31, 32..63; padding rows are zero) hoist out.
  {
    const _Float16* bbase = &smBt[(size_t)(wave * 16 + lm) * LP + lg * 16];
    const v16h b0 = frag_b(bbase);
    const v16h b1 = frag_b(bbase + 32);
#pragma unroll 3
    for (int mt = 0; mt < RB / 16; ++mt) {
      const _Float16* abase = &smW[(size_t)(mt * 16 + lm) * LP + lg * 16];
      const v16h a0 = frag_a(abase);
      const v16h a1 = frag_a(abase + 32);
      v8f acc = {};
      acc = WMMA16(a0, b0, acc);
      acc = WMMA16(a1, b1, acc);
      // C layout: col = lane&15, row = (lane>>4)*8 + j ; relu + f16 store
      const int col = wave * 16 + lm;
      const int rb  = mt * 16 + lg * 8;
#pragma unroll
      for (int j = 0; j < 8; ++j)
        smK[(size_t)(rb + j) * CD + col] = (_Float16)fmaxf(acc[j], 0.0f);
    }
  }
  __syncthreads();

  // ---------------- Stage B: out(32x32) = smK(32,1152) @ win(1152,32) ------
  // Contraction index c = k*128 + d: 36 K-steps of 32; each 32-wide step sits
  // inside one k, so A comes from row (o*9+k) of smK and B from row (t+k) of
  // smP — both as contiguous ds_load_b128 pairs. Split-K: wave w owns tile
  // (w>>1) and K-half (w&1), 18 WMMA each; partials reduced through LDS.
  {
    const int jt = wave >> 1;               // tile index 0..3
    const int mo = jt >> 1;                 // o-tile
    const int nt = jt & 1;                  // t-tile
    const int s0 = (wave & 1) * 18;         // K-half start
    v8f acc = {};
#pragma unroll 3
    for (int s = s0; s < s0 + 18; ++s) {
      const int c0 = s * 32;
      const int kq = c0 >> 7;               // which k this step lies in
      const int d0 = c0 & 127;              // d base within that k
      const _Float16* ap =
          &smK[(size_t)((mo * 16 + lm) * CK + kq) * CD + d0 + lg * 16];
      const _Float16* bp =
          &smP[(size_t)(nt * 16 + lm + kq) * CD + d0 + lg * 16];
      acc = WMMA16(frag_a(ap), frag_b(bp), acc);
    }
    // park partials (C-layout preserved) in the dead smW region
    *(v8f*)&scr[(size_t)wave * 256 + lane * 8] = acc;
  }
  __syncthreads();

  // Pairwise reduce + bias + global store: waves 0..3, one tile each.
  if (wave < 4) {
    const int mo = wave >> 1;
    const int nt = wave & 1;
    const v8f p0 = *(const v8f*)&scr[(size_t)(2 * wave + 0) * 256 + lane * 8];
    const v8f p1 = *(const v8f*)&scr[(size_t)(2 * wave + 1) * 256 + lane * 8];
    // C layout: t = nt*16 + (lane&15), o = obase + mo*16 + (lane>>4)*8 + j
    const int t   = nt * 16 + lm;
    const int ob0 = obase + mo * 16 + lg * 8;
#pragma unroll
    for (int j = 0; j < 8; ++j) {
      const int o = ob0 + j;
      out[((size_t)n * CO + o) * CT + t] = p0[j] + p1[j] + bias[o];
    }
  }
}

extern "C" void kernel_launch(void* const* d_in, const int* in_sizes, int n_in,
                              void* d_out, int out_size, void* d_ws, size_t ws_size,
                              hipStream_t stream) {
  (void)in_sizes; (void)n_in; (void)out_size; (void)d_ws; (void)ws_size;
  const float* pep  = (const float*)d_in[0];
  const float* mhc  = (const float*)d_in[1];
  const float* wgt  = (const float*)d_in[2];
  const float* bias = (const float*)d_in[3];
  float* out = (float*)d_out;

  const size_t shmem = (size_t)SM_TOTAL * sizeof(unsigned short);  // 137216 B
  // Raise the dynamic-LDS cap (gfx1250 WGP has 320 KB); idempotent, capture-safe.
  (void)hipFuncSetAttribute((const void*)iconv_fused_kernel,
                            hipFuncAttributeMaxDynamicSharedMemorySize,
                            (int)shmem);

  dim3 grid(CN * (CO / OB));   // 512 * 8 = 4096 workgroups
  dim3 block(256);             // 8 wave32
  iconv_fused_kernel<<<grid, block, shmem, stream>>>(pep, mhc, wgt, bias, out);
}
